// MutliHeadAttention1D_31585189495126
// MI455X (gfx1250) — matrix operand order repaired and checked by reading, not compiled, not run on hardware
//
#include <hip/hip_runtime.h>
#include <hip/hip_bf16.h>

// ---------------- problem constants ----------------
#define B_    4
#define S_    4096
#define FIN   512
#define FOUT  512
#define KWIN  17
#define GRP   8
#define DH    64          // FOUT / GRP
#define PADW  8           // (KWIN-1)/2
#define M_    (B_ * S_)   // 16384 rows
#define NTOT  (3 * FOUT)  // 1536 fused output channels (q,k,v)
#define KDIM  512

// ---------------- WMMA vector types ----------------
typedef __attribute__((ext_vector_type(16))) __bf16 v16bf;
typedef __attribute__((ext_vector_type(8)))  __bf16 v8bf;
typedef __attribute__((ext_vector_type(8)))  float  v8f;

// LDS tile row stride (elements): 32 data + 8 pad to spread banks
#define LDS_STRIDE 40
#define TILE_BYTES (128 * LDS_STRIDE * 2)   // one A or B buffer, bytes

// Raw LDS byte offset of a __shared__ object (generic -> addrspace(3) -> int)
__device__ __forceinline__ unsigned lds_base(const void* p) {
  return (unsigned)(unsigned long long)(
      (const __attribute__((address_space(3))) unsigned char*)p);
}

// ---------------- fp32 -> bf16 (RNE) ----------------
__global__ __launch_bounds__(256) void cvt_bf16(const float* __restrict__ src,
                                                unsigned short* __restrict__ dst,
                                                int n) {
  int i = blockIdx.x * 256 + threadIdx.x;
  if (i < n) {
    unsigned int u = __float_as_uint(src[i]);
    unsigned int r = u + 0x7FFFu + ((u >> 16) & 1u);
    dst[i] = (unsigned short)(r >> 16);
  }
}

// ---------------- fused QKV projection GEMM ----------------
// C[M=16384, N=1536] = Xbf16[M,512] * Wbf16[N,512]^T, fp32 accumulate.
// 256 threads = 8 waves; tile 128(M) x 128(N); wave = 32x64 subtile (2x4 WMMA).
// Global->LDS via GLOBAL_LOAD_ASYNC_TO_LDS_B128 (ASYNCcnt), double-buffered.
__global__ __launch_bounds__(256) void qkv_gemm(const unsigned short* __restrict__ xb_us,
                                                const unsigned short* __restrict__ wb_us,
                                                float* __restrict__ qkv) {
  __shared__ __bf16 As[2][128 * LDS_STRIDE];
  __shared__ __bf16 Bs[2][128 * LDS_STRIDE];

  const int tid  = threadIdx.x;
  const int lane = tid & 31;
  const int wave = tid >> 5;
  const int half = lane >> 4;   // 0: lanes 0-15, 1: lanes 16-31
  const int l16  = lane & 15;
  const int wm   = wave & 3;    // 4 M-subtiles of 32 rows
  const int wn   = wave >> 2;   // 2 N-subtiles of 64 cols
  const int m0   = blockIdx.x * 128;
  const int n0   = blockIdx.y * 128;

  v8f acc[2][4] = {};

  // copy assignment: each thread moves 32B (2 x b128) of A and of B per slab
  const int ldr = tid >> 1;          // 0..127 tile row
  const int ldc = (tid & 1) * 16;    // 0 or 16 within the 32-wide K slab

  const unsigned ldsA = lds_base(&As[0][0]) + (unsigned)(ldr * LDS_STRIDE + ldc) * 2u;
  const unsigned ldsB = lds_base(&Bs[0][0]) + (unsigned)(ldr * LDS_STRIDE + ldc) * 2u;
  const unsigned gA0  = (unsigned)(((size_t)(m0 + ldr) * KDIM + ldc) * 2u); // byte off
  const unsigned gB0  = (unsigned)(((size_t)(n0 + ldr) * KDIM + ldc) * 2u);

#define ISSUE_SLAB(kb, buf)                                                    \
  do {                                                                         \
    unsigned la = ldsA + (unsigned)(buf) * TILE_BYTES;                         \
    unsigned lb = ldsB + (unsigned)(buf) * TILE_BYTES;                         \
    unsigned ga = gA0 + (unsigned)(kb) * 2u;                                   \
    unsigned gb = gB0 + (unsigned)(kb) * 2u;                                   \
    asm volatile("global_load_async_to_lds_b128 %0, %1, %2"                    \
                 :: "v"(la), "v"(ga), "s"(xb_us) : "memory");                  \
    asm volatile("global_load_async_to_lds_b128 %0, %1, %2 offset:16"         \
                 :: "v"(la), "v"(ga), "s"(xb_us) : "memory");                  \
    asm volatile("global_load_async_to_lds_b128 %0, %1, %2"                    \
                 :: "v"(lb), "v"(gb), "s"(wb_us) : "memory");                  \
    asm volatile("global_load_async_to_lds_b128 %0, %1, %2 offset:16"         \
                 :: "v"(lb), "v"(gb), "s"(wb_us) : "memory");                  \
  } while (0)

  // prologue: fill buffer 0
  ISSUE_SLAB(0, 0);
  asm volatile("s_wait_asynccnt 0x0" ::: "memory");
  __syncthreads();

  for (int kb = 0; kb < KDIM; kb += 32) {
    const int cur = (kb >> 5) & 1;
    if (kb + 32 < KDIM) ISSUE_SLAB(kb + 32, cur ^ 1);  // overlap next slab

    // A fragments: 16x32 bf16 (ISA 7.12.2): lane half selects K-octets
    v16bf afrag[2];
#pragma unroll
    for (int mt = 0; mt < 2; ++mt) {
      const int row = wm * 32 + mt * 16 + l16;
      const v8bf lo = *(const v8bf*)(&As[cur][row * LDS_STRIDE + half * 8]);
      const v8bf hi = *(const v8bf*)(&As[cur][row * LDS_STRIDE + 16 + half * 8]);
#pragma unroll
      for (int i = 0; i < 8; ++i) { afrag[mt][i] = lo[i]; afrag[mt][8 + i] = hi[i]; }
    }
    // B fragments: 32x16 bf16: lane half selects K-octets (same order as A)
    v16bf bfrag[4];
#pragma unroll
    for (int nt = 0; nt < 4; ++nt) {
      const int col = wn * 64 + nt * 16 + l16;
      const v8bf lo = *(const v8bf*)(&Bs[cur][col * LDS_STRIDE + half * 8]);
      const v8bf hi = *(const v8bf*)(&Bs[cur][col * LDS_STRIDE + 16 + half * 8]);
#pragma unroll
      for (int i = 0; i < 8; ++i) { bfrag[nt][i] = lo[i]; bfrag[nt][8 + i] = hi[i]; }
    }

#pragma unroll
    for (int mt = 0; mt < 2; ++mt)
#pragma unroll
      for (int nt = 0; nt < 4; ++nt)
        acc[mt][nt] = __builtin_amdgcn_wmma_f32_16x16x32_bf16(
            false, afrag[mt], false, bfrag[nt],
            (short)0, acc[mt][nt], false, false);

    // my async writes done + everyone's LDS reads of `cur` retired -> safe swap
    asm volatile("s_wait_asynccnt 0x0" ::: "memory");
    __syncthreads();
  }
#undef ISSUE_SLAB

  // Store: D layout, VGPR r -> M = r + 8*half, N = l16 within each 16x16 tile.
#pragma unroll
  for (int mt = 0; mt < 2; ++mt) {
    const int mbase = m0 + wm * 32 + mt * 16 + half * 8;
#pragma unroll
    for (int nt = 0; nt < 4; ++nt) {
      const int ng   = n0 + wn * 64 + nt * 16 + l16;
      const int proj = ng >> 9;       // 0=q,1=k,2=v (512-channel blocks)
      const int o    = ng & 511;
      float* dst = qkv + (size_t)proj * M_ * FOUT + o;
#pragma unroll
      for (int r = 0; r < 8; ++r)
        dst[(size_t)(mbase + r) * FOUT] = acc[mt][nt][r];
    }
  }
}

// ---------------- windowed attention ----------------
// One wave32 per (b,s,g). Lanes own 2 consecutive head channels (DH=64).
// Padding: positions outside [0,S) contribute k=0 (energy term is q·rel) and v=0.
__global__ __launch_bounds__(256) void attn_kernel(const float* __restrict__ qkv,
                                                   const float* __restrict__ rel,
                                                   float* __restrict__ out,
                                                   float* __restrict__ attn) {
  const float* __restrict__ qb = qkv;
  const float* __restrict__ kb = qkv + (size_t)M_ * FOUT;
  const float* __restrict__ vb = qkv + 2 * (size_t)M_ * FOUT;

  const int lane = threadIdx.x & 31;
  const int wid  = blockIdx.x * 8 + (threadIdx.x >> 5);
  const int g = wid & (GRP - 1);
  const int s = (wid >> 3) & (S_ - 1);
  const int b = wid >> 15;            // S_*GRP = 2^15
  const int row = b * S_ + s;
  const int o0  = g * DH + lane * 2;  // channel pair for this lane

  const float2 q2 = *(const float2*)(qb + (size_t)row * FOUT + o0);

  float e[KWIN];
#pragma unroll
  for (int k = 0; k < KWIN; ++k) {
    const int p = s + k - PADW;
    float2 kv = make_float2(0.f, 0.f);
    if (p >= 0 && p < S_)
      kv = *(const float2*)(kb + (size_t)(b * S_ + p) * FOUT + o0);
    const float r0 = rel[o0 * KWIN + k];
    const float r1 = rel[(o0 + 1) * KWIN + k];
    float part = q2.x * (kv.x + r0) + q2.y * (kv.y + r1);
#pragma unroll
    for (int off = 16; off >= 1; off >>= 1)
      part += __shfl_xor(part, off, 32);
    e[k] = part;  // wave-uniform
  }

  float mx = e[0];
#pragma unroll
  for (int k = 1; k < KWIN; ++k) mx = fmaxf(mx, e[k]);
  float sum = 0.f;
#pragma unroll
  for (int k = 0; k < KWIN; ++k) { e[k] = __expf(e[k] - mx); sum += e[k]; }
  const float inv = 1.0f / sum;

  float2 acc = make_float2(0.f, 0.f);
#pragma unroll
  for (int k = 0; k < KWIN; ++k) {
    e[k] *= inv;
    const int p = s + k - PADW;
    if (p >= 0 && p < S_) {
      const float2 vv = *(const float2*)(vb + (size_t)(b * S_ + p) * FOUT + o0);
      acc.x += e[k] * vv.x;
      acc.y += e[k] * vv.y;
    }
  }

  *(float2*)(out + (size_t)row * FOUT + o0) = acc;

  if (lane == 0) {
    float* ap = attn + ((size_t)row * GRP + g) * KWIN;
#pragma unroll
    for (int k = 0; k < KWIN; ++k) ap[k] = e[k];
  }
}

// ---------------- launcher ----------------
extern "C" void kernel_launch(void* const* d_in, const int* in_sizes, int n_in,
                              void* d_out, int out_size, void* d_ws, size_t ws_size,
                              hipStream_t stream) {
  const float* x   = (const float*)d_in[0];
  const float* Wq  = (const float*)d_in[1];
  const float* Wk  = (const float*)d_in[2];
  const float* Wv  = (const float*)d_in[3];
  const float* rel = (const float*)d_in[4];

  // workspace layout:
  //   xb  : M_*KDIM bf16          (16 MB)
  //   wb  : NTOT*KDIM bf16        (1.5 MB)   q,k,v weights concatenated
  //   qkv : 3*M_*FOUT fp32        (96 MB)
  char* ws = (char*)d_ws;
  unsigned short* xb = (unsigned short*)ws;
  unsigned short* wb = (unsigned short*)(ws + (size_t)M_ * KDIM * 2);
  float* qkv = (float*)(ws + (size_t)M_ * KDIM * 2 + (size_t)NTOT * KDIM * 2);

  cvt_bf16<<<(M_ * KDIM) / 256, 256, 0, stream>>>(x, xb, M_ * KDIM);
  cvt_bf16<<<(FOUT * FIN) / 256, 256, 0, stream>>>(Wq, wb, FOUT * FIN);
  cvt_bf16<<<(FOUT * FIN) / 256, 256, 0, stream>>>(Wk, wb + FOUT * FIN, FOUT * FIN);
  cvt_bf16<<<(FOUT * FIN) / 256, 256, 0, stream>>>(Wv, wb + 2 * FOUT * FIN, FOUT * FIN);

  dim3 gg(M_ / 128, NTOT / 128);   // 128 x 12
  qkv_gemm<<<gg, 256, 0, stream>>>(xb, wb, qkv);

  float* outp  = (float*)d_out;
  float* attnp = outp + (size_t)M_ * FOUT;
  attn_kernel<<<(B_ * S_ * GRP) / 8, 256, 0, stream>>>(qkv, rel, outp, attnp);
}
